// PointerNetwork_23003844837742
// MI455X (gfx1250) — matrix operand order repaired and checked by reading, not compiled, not run on hardware
//
#include <hip/hip_runtime.h>
#include <hip/hip_bf16.h>
#include <stdint.h>

// ---------------------------------------------------------------------------
// PointerNetwork on MI455X (gfx1250, wave32, WMMA, async global->LDS).
//   1. relayout x -> bf16 [s][b][e]; transpose weights -> bf16 [N,K]
//   2. WMMA GEMMs (bf16 in / bf16 or f32 out)
//   3. persistent-workgroup encoder scan (fwd+bwd, grid=2) with
//      double-buffered global_load_async_to_lds_b128 slab prefetch
//   4. decoder-input + attention-bias GEMMs
//   5. persistent decoder scan (same async pipeline)
//   6. q = hdec@W4
//   7. fused tanh-score + masked argmax (native v_tanh_f32 when available;
//      out2/vt2/chunk_emb provably cancel in the argmax)
//   8. serial bond chain -> one-hot [S,B]
// ---------------------------------------------------------------------------

typedef __bf16 bf16;
typedef bf16  v16bf __attribute__((ext_vector_type(16)));
typedef bf16  v8bf  __attribute__((ext_vector_type(8)));
typedef float v8f   __attribute__((ext_vector_type(8)));

static constexpr int BB = 16, SS = 1024, EE = 256, HH = 256, WT = 256;
static constexpr int FOURH = 4 * HH;   // 1024
static constexpr int KW = 256;

__device__ __forceinline__ float sigm(float x) { return 1.0f / (1.0f + __expf(-x)); }
__device__ __forceinline__ float tanh_fast(float x) {
#if __has_builtin(__builtin_amdgcn_tanhf)
  return __builtin_amdgcn_tanhf(x);          // v_tanh_f32 (gfx1250 TRANS op)
#else
  return 1.0f - 2.0f / (__expf(2.0f * x) + 1.0f);
#endif
}

// LDS byte offset of a generic pointer into shared memory (AS3 is 32-bit).
typedef __attribute__((address_space(3))) const char* lds_cptr_t;
__device__ __forceinline__ uint32_t lds_off_u32(const void* p) {
  return (uint32_t)(uintptr_t)(lds_cptr_t)p;
}
// Async DMA: 16B per lane, global -> LDS, tracked by ASYNCcnt.
__device__ __forceinline__ void async_copy_b128(uint32_t ldsOff, const void* g) {
  asm volatile("global_load_async_to_lds_b128 %0, %1, off"
               :: "v"(ldsOff), "v"(g) : "memory");
}
__device__ __forceinline__ void wait_async_le8() {
  asm volatile("s_wait_asynccnt 0x8" ::: "memory");
}
__device__ __forceinline__ void wait_async_0() {
  asm volatile("s_wait_asynccnt 0x0" ::: "memory");
}

// A fragment, 16x32 bf16 (MxK): lane=M%16 (+16 => K-hi); 16 contiguous halves
__device__ __forceinline__ v16bf load_fragA(const bf16* p) {
  union { v16bf v; v8bf h[2]; } f;
  f.h[0] = *(const v8bf*)(p);
  f.h[1] = *(const v8bf*)(p + 8);
  return f.v;
}
// B fragment, 32x16 bf16 from B^T [N,K]: lane=N%16 (+16 => K-hi); 16 contiguous halves
__device__ __forceinline__ v16bf load_fragB(const bf16* p) {
  union { v16bf v; v8bf h[2]; } f;
  f.h[0] = *(const v8bf*)(p);
  f.h[1] = *(const v8bf*)(p + 8);
  return f.v;
}

// --------------------------- data prep kernels -----------------------------

__global__ void conv_x_kernel(const float* __restrict__ x, bf16* __restrict__ xb) {
  const int total = BB * SS * EE;
  for (int i = blockIdx.x * blockDim.x + threadIdx.x; i < total; i += gridDim.x * blockDim.x) {
    const int e = i & (EE - 1);
    const int r = i >> 8;
    const int b = r & (BB - 1);
    const int s = r >> 4;
    xb[i] = (bf16)x[((size_t)b * SS + s) * EE + e];
  }
}

__global__ void wconv_t_kernel(const float* __restrict__ W, bf16* __restrict__ out,
                               int N, int rowOff) {
  const int total = N * KW;
  for (int i = blockIdx.x * blockDim.x + threadIdx.x; i < total; i += gridDim.x * blockDim.x) {
    const int k = i & (KW - 1);
    const int n = i >> 8;
    out[i] = (bf16)W[(size_t)(k + rowOff) * N + n];
  }
}

__global__ void zero_f32_kernel(float* p, int n) {
  const int i = blockIdx.x * blockDim.x + threadIdx.x;
  if (i < n) p[i] = 0.0f;
}

// ----------------------------- WMMA GEMM -----------------------------------
// C (+)= A[M,K]bf16 @ Bt[N,K]bf16.  8 waves; wave tile 64x32; block 64x256.
// Output/accumulator storage: f32 (Cf) or bf16 (Cb) selected by pointer.
__global__ __launch_bounds__(256) void gemm_bf16_nt(
    const bf16* __restrict__ A, const bf16* __restrict__ Bt,
    float* __restrict__ Cf, bf16* __restrict__ Cb,
    int M, int N, int K, int accflag) {
  const int wave = threadIdx.x >> 5;
  const int lane = threadIdx.x & 31;
  const int lm = lane & 15, hi = lane >> 4;
  const int m0 = blockIdx.x * 64;
  const int n0 = blockIdx.y * 256 + wave * 32;
  const bool outBf = (Cb != nullptr);

  v8f acc[4][2];
  for (int i = 0; i < 4; i++)
    for (int j = 0; j < 2; j++) {
      const size_t base = (size_t)(m0 + i * 16 + 8 * hi) * N + (n0 + j * 16 + lm);
      if (accflag) {
        if (outBf) for (int g = 0; g < 8; g++) acc[i][j][g] = (float)Cb[base + (size_t)g * N];
        else       for (int g = 0; g < 8; g++) acc[i][j][g] = Cf[base + (size_t)g * N];
      } else {
        for (int g = 0; g < 8; g++) acc[i][j][g] = 0.0f;
      }
    }

  for (int k0 = 0; k0 < K; k0 += 32) {
    v16bf a[4], b[2];
    for (int i = 0; i < 4; i++) {
      const bf16* pa = A + (size_t)(m0 + i * 16 + lm) * K + k0 + hi * 16;
      __builtin_prefetch(pa + K, 0, 3);   // WGP-scope global_prefetch_b8
      a[i] = load_fragA(pa);
    }
    for (int j = 0; j < 2; j++)
      b[j] = load_fragB(Bt + (size_t)(n0 + j * 16 + lm) * K + k0 + hi * 16);
    for (int i = 0; i < 4; i++)
      for (int j = 0; j < 2; j++)
        acc[i][j] = __builtin_amdgcn_wmma_f32_16x16x32_bf16(
            false, a[i], false, b[j], (short)0, acc[i][j], false, false);
  }

  for (int i = 0; i < 4; i++)
    for (int j = 0; j < 2; j++) {
      const size_t base = (size_t)(m0 + i * 16 + 8 * hi) * N + (n0 + j * 16 + lm);
      if (outBf) for (int g = 0; g < 8; g++) Cb[base + (size_t)g * N] = (bf16)acc[i][j][g];
      else       for (int g = 0; g < 8; g++) Cf[base + (size_t)g * N] = acc[i][j][g];
    }
}

// ------------------------- persistent LSTM scans ---------------------------
// 8 waves.  Wave w owns gate columns [w*32,w*32+32) of each of i,f,g,o ->
// gate math fully in-register; c lives in accumulator VGPRs; h round-trips
// through 8KB LDS bf16.  Next step's 32KB z-slab is DMA'd into a double
// buffer with global_load_async_to_lds_b128 while the current step computes.

static constexpr int SLAB = BB * FOURH;            // 16384 bf16 = 32KB
static constexpr int SLAB_BYTES = SLAB * 2;

__device__ __forceinline__ void issue_slab(const bf16* zpre, int srow,
                                           bf16* buf, int tid) {
  const char* g = (const char*)(zpre + (size_t)srow * FOURH);
  const uint32_t l0 = lds_off_u32(buf);
  for (int q = 0; q < 8; q++)
    async_copy_b128(l0 + q * 4096 + tid * 16, g + q * 4096 + tid * 16);
}

__global__ __launch_bounds__(256) void encoder_kernel(
    const bf16* __restrict__ zF, const float* __restrict__ bF,
    const bf16* __restrict__ wF, bf16* __restrict__ oF,
    const bf16* __restrict__ zB, const float* __restrict__ bB,
    const bf16* __restrict__ wB, bf16* __restrict__ oB) {
  const int dir = blockIdx.x;            // 0 fwd, 1 bwd
  const bf16* zpre  = dir ? zB : zF;     // x@Wih (bf16) [s*16+b][1024]
  const float* bias = dir ? bB : bF;
  const bf16* WhhT  = dir ? wB : wF;     // [1024,256] bf16
  bf16* out         = dir ? oB : oF;

  __shared__ bf16 zsh[2][SLAB];
  __shared__ bf16 hsh[BB * HH];
  const int tid = threadIdx.x;
  const int wave = tid >> 5, lane = tid & 31;
  const int lm = lane & 15, hi = lane >> 4;

  for (int i = tid; i < BB * HH; i += 256) hsh[i] = (bf16)0.0f;
  v8f c[2];
  for (int u = 0; u < 2; u++)
    for (int g = 0; g < 8; g++) c[u][g] = 0.0f;

  issue_slab(zpre, (dir ? (SS - 1) : 0) * BB, &zsh[0][0], tid);  // slab(0)

  __syncthreads();
  for (int step = 0; step < SS; step++) {
    const int s = dir ? (SS - 1 - step) : step;
    if (step + 1 < SS) {
      const int sn = dir ? (SS - 2 - step) : (step + 1);
      issue_slab(zpre, sn * BB, &zsh[(step + 1) & 1][0], tid);
      wait_async_le8();                 // in-order: slab(step) retired
    } else {
      wait_async_0();
    }
    __syncthreads();                    // everyone's slab writes visible

    const bf16* zrow = &zsh[step & 1][0];
    v8f z[4][2];
    for (int gc = 0; gc < 4; gc++)
      for (int u = 0; u < 2; u++) {
        const int n = gc * HH + wave * 32 + u * 16 + lm;
        const float bv = bias[n];
        for (int g = 0; g < 8; g++)
          z[gc][u][g] = (float)zrow[(g + 8 * hi) * FOURH + n] + bv;
      }
    for (int k0 = 0; k0 < HH; k0 += 32) {
      const v16bf a = load_fragA(hsh + lm * HH + k0 + hi * 16);
      for (int gc = 0; gc < 4; gc++)
        for (int u = 0; u < 2; u++) {
          const int n = gc * HH + wave * 32 + u * 16 + lm;
          const v16bf bfr = load_fragB(WhhT + (size_t)n * HH + k0 + hi * 16);
          z[gc][u] = __builtin_amdgcn_wmma_f32_16x16x32_bf16(
              false, a, false, bfr, (short)0, z[gc][u], false, false);
        }
    }
    v8f h2[2];
    for (int u = 0; u < 2; u++)
      for (int g = 0; g < 8; g++) {
        const float iv = sigm(z[0][u][g]);
        const float fv = sigm(z[1][u][g]);
        const float gv = tanh_fast(z[2][u][g]);
        const float ov = sigm(z[3][u][g]);
        const float c2 = fv * c[u][g] + iv * gv;
        c[u][g] = c2;
        h2[u][g] = ov * tanh_fast(c2);
      }
    __syncthreads();                    // all reads of old h done
    for (int u = 0; u < 2; u++)
      for (int g = 0; g < 8; g++) {
        const int bIdx = g + 8 * hi;
        const int n = wave * 32 + u * 16 + lm;
        const bf16 hv = (bf16)h2[u][g];
        hsh[bIdx * HH + n] = hv;
        out[((size_t)s * BB + bIdx) * HH + n] = hv;
      }
    __syncthreads();                    // new h visible
  }
}

__global__ __launch_bounds__(256) void decoder_kernel(
    const bf16* __restrict__ zpre /* decInW bf16 */, const float* __restrict__ bias,
    const bf16* __restrict__ WhhT, const bf16* __restrict__ hn_bf,
    bf16* __restrict__ out /* hdec bf16 */) {
  __shared__ bf16 zsh[2][SLAB];
  __shared__ bf16 hsh[BB * HH];
  const int tid = threadIdx.x;
  const int wave = tid >> 5, lane = tid & 31;
  const int lm = lane & 15, hi = lane >> 4;

  for (int i = tid; i < BB * HH; i += 256) hsh[i] = (bf16)0.0f;
  v8f c[2];
  for (int u = 0; u < 2; u++)
    for (int g = 0; g < 8; g++) {
      const int bIdx = g + 8 * hi;
      const int n = wave * 32 + u * 16 + lm;
      c[u][g] = (float)hn_bf[((size_t)(SS - 1) * BB + bIdx) * HH + n]; // c0 = hn[-1]
    }
  __syncthreads();

  // slab(t) = decInW rows (t-1)*16, consumed at step t (t>=1), parity t&1
  for (int t = 0; t < SS; t++) {
    if (t + 1 < SS) {
      issue_slab(zpre, t * BB, &zsh[(t + 1) & 1][0], tid);
      wait_async_le8();
    } else {
      wait_async_0();
    }
    __syncthreads();

    v8f z[4][2];
    const bf16* zrow = &zsh[t & 1][0];
    for (int gc = 0; gc < 4; gc++)
      for (int u = 0; u < 2; u++) {
        const int n = gc * HH + wave * 32 + u * 16 + lm;
        const float bv = bias[n];
        if (t > 0) {
          for (int g = 0; g < 8; g++)
            z[gc][u][g] = (float)zrow[(g + 8 * hi) * FOURH + n] + bv;
        } else {
          for (int g = 0; g < 8; g++) z[gc][u][g] = bv;   // zero input at t==0
        }
      }
    for (int k0 = 0; k0 < HH; k0 += 32) {
      const v16bf a = load_fragA(hsh + lm * HH + k0 + hi * 16);
      for (int gc = 0; gc < 4; gc++)
        for (int u = 0; u < 2; u++) {
          const int n = gc * HH + wave * 32 + u * 16 + lm;
          const v16bf bfr = load_fragB(WhhT + (size_t)n * HH + k0 + hi * 16);
          z[gc][u] = __builtin_amdgcn_wmma_f32_16x16x32_bf16(
              false, a, false, bfr, (short)0, z[gc][u], false, false);
        }
    }
    v8f h2[2];
    for (int u = 0; u < 2; u++)
      for (int g = 0; g < 8; g++) {
        const float iv = sigm(z[0][u][g]);
        const float fv = sigm(z[1][u][g]);
        const float gv = tanh_fast(z[2][u][g]);
        const float ov = sigm(z[3][u][g]);
        const float c2 = fv * c[u][g] + iv * gv;
        c[u][g] = c2;
        h2[u][g] = ov * tanh_fast(c2);
      }
    __syncthreads();
    for (int u = 0; u < 2; u++)
      for (int g = 0; g < 8; g++) {
        const int bIdx = g + 8 * hi;
        const int n = wave * 32 + u * 16 + lm;
        const bf16 hv = (bf16)h2[u][g];
        hsh[bIdx * HH + n] = hv;
        out[((size_t)t * BB + bIdx) * HH + n] = hv;
      }
    __syncthreads();
  }
}

// ---------------- fused attention scoring + masked argmax ------------------
__global__ __launch_bounds__(256) void attn_argmax_kernel(
    const float* __restrict__ pre, const float* __restrict__ b3,
    const float* __restrict__ q, const float* __restrict__ vt1,
    int* __restrict__ rmax) {
  const int t = blockIdx.x, b = blockIdx.y;
  __shared__ float m3[WT];
  __shared__ float v1[WT];
  __shared__ float rs[256];
  __shared__ int rj[256];
  const int tid = threadIdx.x;
  const size_t rowTB = (size_t)(t * BB + b) * WT;
  m3[tid] = b3[rowTB + tid] + q[rowTB + tid];
  v1[tid] = vt1[tid];
  __syncthreads();

  float best = -3.4e38f;
  int bestj = SS;
  for (int j = t + tid; j < SS; j += 256) {
    const float* p = pre + (size_t)(j * BB + b) * WT;
    float s = 0.0f;
    for (int w = 0; w < WT; w++) s += tanh_fast(p[w] + m3[w]) * v1[w];
    if (s > best) { best = s; bestj = j; }
  }
  rs[tid] = best; rj[tid] = bestj;
  __syncthreads();
  for (int off = 128; off > 0; off >>= 1) {
    if (tid < off) {
      const float so = rs[tid + off];
      const int jo = rj[tid + off];
      if (so > rs[tid] || (so == rs[tid] && jo < rj[tid])) { rs[tid] = so; rj[tid] = jo; }
    }
    __syncthreads();
  }
  if (tid == 0) rmax[t * BB + b] = rj[0];
}

__global__ void finalize_chain_kernel(const int* __restrict__ rmax, float* __restrict__ out) {
  const int b = threadIdx.x;
  if (b >= BB) return;
  int bond = 0;
  for (int t = 0; t < SS; t++) {
    if (t == bond) {
      const int r = rmax[t * BB + b];
      out[(size_t)r * BB + b] = 1.0f;
      bond = r;
    }
  }
}

// ---------------------------------------------------------------------------

extern "C" void kernel_launch(void* const* d_in, const int* in_sizes, int n_in,
                              void* d_out, int out_size, void* d_ws, size_t ws_size,
                              hipStream_t stream) {
  (void)in_sizes; (void)n_in; (void)out_size; (void)ws_size;
  const float* x       = (const float*)d_in[0];
  const float* encWihF = (const float*)d_in[1];
  const float* encWhhF = (const float*)d_in[2];
  const float* encBF   = (const float*)d_in[3];
  const float* encWihB = (const float*)d_in[4];
  const float* encWhhB = (const float*)d_in[5];
  const float* encBB   = (const float*)d_in[6];
  const float* decWih  = (const float*)d_in[7];
  const float* decWhh  = (const float*)d_in[8];
  const float* decBias = (const float*)d_in[9];
  const float* W1      = (const float*)d_in[10];
  const float* W2      = (const float*)d_in[11];
  const float* W3      = (const float*)d_in[12];
  const float* W4      = (const float*)d_in[13];
  const float* vt1     = (const float*)d_in[14];
  // d_in[15] (vt2), d_in[16] (chunk_emb): uniform over j -> cannot affect argmax.

  char* ws = (char*)d_ws;
  size_t off = 0;
  auto alloc = [&](size_t bytes) -> char* {
    char* p = ws + off;
    off = (off + bytes + 255) & ~(size_t)255;
    return p;
  };

  bf16* xBF   = (bf16*)alloc((size_t)BB * SS * EE * 2);
  bf16* wihFT = (bf16*)alloc((size_t)FOURH * KW * 2);
  bf16* wihBT = (bf16*)alloc((size_t)FOURH * KW * 2);
  bf16* whhFT = (bf16*)alloc((size_t)FOURH * KW * 2);
  bf16* whhBT = (bf16*)alloc((size_t)FOURH * KW * 2);
  bf16* dwxT  = (bf16*)alloc((size_t)FOURH * KW * 2);
  bf16* dwhT  = (bf16*)alloc((size_t)FOURH * KW * 2);
  bf16* dwhhT = (bf16*)alloc((size_t)FOURH * KW * 2);
  bf16* w1aT  = (bf16*)alloc((size_t)WT * KW * 2);
  bf16* w1bT  = (bf16*)alloc((size_t)WT * KW * 2);
  bf16* w2T   = (bf16*)alloc((size_t)WT * KW * 2);
  bf16* w3T   = (bf16*)alloc((size_t)WT * KW * 2);
  bf16* w4T   = (bf16*)alloc((size_t)WT * KW * 2);
  bf16* xWF   = (bf16*)alloc((size_t)BB * SS * FOURH * 2);   // bf16 now
  bf16* xWB   = (bf16*)alloc((size_t)BB * SS * FOURH * 2);
  bf16* dInW  = (bf16*)alloc((size_t)BB * SS * FOURH * 2);
  float* preB = (float*)alloc((size_t)BB * SS * WT * 4);
  float* b3B  = (float*)alloc((size_t)BB * SS * WT * 4);
  float* qB   = (float*)alloc((size_t)BB * SS * WT * 4);
  bf16* hnBF  = (bf16*)alloc((size_t)BB * SS * HH * 2);
  bf16* hbBF  = (bf16*)alloc((size_t)BB * SS * HH * 2);
  bf16* hdBF  = (bf16*)alloc((size_t)BB * SS * HH * 2);
  int* rmax   = (int*)alloc((size_t)SS * BB * 4);

  // 1) relayout / convert
  conv_x_kernel<<<1024, 256, 0, stream>>>(x, xBF);
  wconv_t_kernel<<<1024, 256, 0, stream>>>(encWihF, wihFT, FOURH, 0);
  wconv_t_kernel<<<1024, 256, 0, stream>>>(encWihB, wihBT, FOURH, 0);
  wconv_t_kernel<<<1024, 256, 0, stream>>>(encWhhF, whhFT, FOURH, 0);
  wconv_t_kernel<<<1024, 256, 0, stream>>>(encWhhB, whhBT, FOURH, 0);
  wconv_t_kernel<<<1024, 256, 0, stream>>>(decWih, dwxT, FOURH, 0);
  wconv_t_kernel<<<1024, 256, 0, stream>>>(decWih, dwhT, FOURH, 256);
  wconv_t_kernel<<<1024, 256, 0, stream>>>(decWhh, dwhhT, FOURH, 0);
  wconv_t_kernel<<<256, 256, 0, stream>>>(W1, w1aT, WT, 0);
  wconv_t_kernel<<<256, 256, 0, stream>>>(W1, w1bT, WT, 256);
  wconv_t_kernel<<<256, 256, 0, stream>>>(W2, w2T, WT, 0);
  wconv_t_kernel<<<256, 256, 0, stream>>>(W3, w3T, WT, 0);
  wconv_t_kernel<<<256, 256, 0, stream>>>(W4, w4T, WT, 0);

  const dim3 gBig(BB * SS / 64, FOURH / 256);
  const dim3 gSm(BB * SS / 64, 1);

  // 2) input-projection GEMMs (bf16 out)
  gemm_bf16_nt<<<gBig, 256, 0, stream>>>(xBF, wihFT, nullptr, xWF, BB * SS, FOURH, EE, 0);
  gemm_bf16_nt<<<gBig, 256, 0, stream>>>(xBF, wihBT, nullptr, xWB, BB * SS, FOURH, EE, 0);

  // 3) bidirectional encoder scan
  encoder_kernel<<<2, 256, 0, stream>>>(xWF, encBF, whhFT, hnBF,
                                        xWB, encBB, whhBT, hbBF);

  // 4) decoder-input (bf16 out) + attention-bias (f32 out) GEMMs
  gemm_bf16_nt<<<gBig, 256, 0, stream>>>(xBF, dwxT, nullptr, dInW, BB * SS, FOURH, EE, 0);
  gemm_bf16_nt<<<gBig, 256, 0, stream>>>(hnBF, dwhT, nullptr, dInW, BB * SS, FOURH, HH, 1);
  gemm_bf16_nt<<<gSm, 256, 0, stream>>>(hnBF, w1aT, preB, nullptr, BB * SS, WT, HH, 0);
  gemm_bf16_nt<<<gSm, 256, 0, stream>>>(hbBF, w1bT, preB, nullptr, BB * SS, WT, HH, 1);
  gemm_bf16_nt<<<gSm, 256, 0, stream>>>(xBF, w2T, preB, nullptr, BB * SS, WT, EE, 1);
  gemm_bf16_nt<<<gSm, 256, 0, stream>>>(xBF, w3T, b3B, nullptr, BB * SS, WT, EE, 0);

  // 5) decoder scan
  decoder_kernel<<<1, 256, 0, stream>>>(dInW, decBias, dwhhT, hnBF, hdBF);

  // 6) query projection
  gemm_bf16_nt<<<gSm, 256, 0, stream>>>(hdBF, w4T, qB, nullptr, BB * SS, WT, HH, 0);

  // 7) + 8) argmax table, then serial bond chain into zeroed output
  zero_f32_kernel<<<(SS * BB + 255) / 256, 256, 0, stream>>>((float*)d_out, SS * BB);
  const dim3 gAtt(SS, BB);
  attn_argmax_kernel<<<gAtt, 256, 0, stream>>>(preB, b3B, qB, vt1, rmax);
  finalize_chain_kernel<<<1, 32, 0, stream>>>(rmax, (float*)d_out);
}
